// HighOrderGNNEncoder_86303072845900
// MI455X (gfx1250) — matrix operand order repaired and checked by reading, not compiled, not run on hardware
//
#include <hip/hip_runtime.h>
#include <hip/hip_bf16.h>

// ---------------- problem constants (match reference) ----------------
#define NN  50000
#define EE  800000
#define DD  128
#define HH  128
#define OO  64
#define BB  64

typedef __attribute__((ext_vector_type(16))) _Float16 v16h;
typedef __attribute__((ext_vector_type(8)))  float    v8f;

// ---------------- init helpers ----------------
__global__ void fill_f32(float* __restrict__ p, float val, long n) {
    long i = (long)blockIdx.x * blockDim.x + threadIdx.x;
    long stride = (long)gridDim.x * blockDim.x;
    for (; i < n; i += stride) p[i] = val;
}

// ---------------- degree: deg[dst] += 1 (deg pre-filled with 1.0 = self loop) ----------------
__global__ void degree_kernel(const int* __restrict__ ei, float* __restrict__ deg, int nE) {
    int e = blockIdx.x * blockDim.x + threadIdx.x;
    if (e < nE) atomicAdd(&deg[ei[nE + e]], 1.0f);
}

// ---------------- GEMM: Hout[N,128] = X[N,128] @ W[128,128]^T via WMMA f16 ----------------
// W is converted f32->f16 into LDS once per block (row-padded to kill bank conflicts).
// Each wave owns a 16-row stripe and computes all 8 column tiles (16x128 output),
// reusing its A fragment across 8 WMMAs per K-step: X is read exactly once.
// B fragments for a whole K-step are staged into registers BEFORE the WMMA burst so
// all 16 ds_load_b128 are in flight concurrently instead of serializing per WMMA.
#define WSTRIDE 136   // 128 + 8 halves pad: row stride 272B -> lanes 0..15 hit distinct banks
__global__ void gemm_xwT_wmma(const float* __restrict__ X, const float* __restrict__ W,
                              float* __restrict__ Hout, int Nrows) {
    __shared__ _Float16 Wl[HH * WSTRIDE];

    // cooperative convert W (128x128 f32) -> padded f16 in LDS
    for (int i = threadIdx.x; i < HH * DD; i += blockDim.x) {
        int row = i >> 7, col = i & 127;
        Wl[row * WSTRIDE + col] = (_Float16)W[i];
    }
    __syncthreads();

    const int lane = threadIdx.x & 31;
    const int wave = threadIdx.x >> 5;
    const int M0   = (blockIdx.x * 8 + wave) * 16;
    if (M0 >= Nrows) return;                    // after the only barrier

    const int r   = lane & 15;                  // A: row in tile, B: column in tile
    const int kbA = (lane < 16) ? 0 : 16;       // A lane-half K base (contiguous 16)
    const int kbB = (lane < 16) ? 0 : 16;       // B lane-half K base (contiguous 16)

    v8f acc[8] = {};

    #pragma unroll
    for (int kk = 0; kk < DD; kk += 32) {
        // stage A fragment (global, f32 -> f16)
        v16h a;
        const float* xrow = X + (size_t)(M0 + r) * DD + kk + kbA;
        #pragma unroll
        for (int e = 0; e < 8; ++e) {
            a[e]     = (_Float16)xrow[e];        // K = kbA + e
            a[e + 8] = (_Float16)xrow[e + 8];    // K = kbA + 8 + e
        }
        // stage all 8 B fragments for this K-step (independent ds_load_b128 pairs)
        v16h bfrag[8];
        #pragma unroll
        for (int ct = 0; ct < 8; ++ct) {
            // B(k, n) = W[ct*16 + n, k]; lane holds column n=r, 16 contiguous K halves
            bfrag[ct] = *(const v16h*)(Wl + (ct * 16 + r) * WSTRIDE + kk + kbB);
        }
        // WMMA burst: same A, 8 independent accumulator streams
        #pragma unroll
        for (int ct = 0; ct < 8; ++ct) {
            acc[ct] = __builtin_amdgcn_wmma_f32_16x16x32_f16(false, a, false, bfrag[ct],
                                                             (short)0, acc[ct], false, false);
        }
    }

    // C/D layout: lanes 0-15 -> N=lane, M=vgpr; lanes 16-31 -> N=lane-16, M=vgpr+8
    const int n    = lane & 15;
    const int moff = (lane < 16) ? 0 : 8;
    #pragma unroll
    for (int ct = 0; ct < 8; ++ct) {
        #pragma unroll
        for (int rr = 0; rr < 8; ++rr) {
            Hout[(size_t)(M0 + moff + rr) * HH + ct * 16 + n] = acc[ct][rr];
        }
    }
}

// ---------------- edge scatter: agg[dst] += h[src] * rsqrt(deg[src])*rsqrt(deg[dst]) ----------------
// One wave per edge; each lane moves 4 consecutive floats (128 = 32 lanes * 4).
__global__ void edge_scatter(const float* __restrict__ Hfeat, const int* __restrict__ ei,
                             const float* __restrict__ deg, float* __restrict__ agg, int nE) {
    int gid  = blockIdx.x * blockDim.x + threadIdx.x;
    int e    = gid >> 5;
    int lane = gid & 31;
    if (e >= nE) return;
    int src = ei[e];
    int dst = ei[nE + e];
    float norm = rsqrtf(deg[src]) * rsqrtf(deg[dst]);
    const float4 hv = *(const float4*)(Hfeat + (size_t)src * HH + lane * 4);
    float* ap = agg + (size_t)dst * HH + lane * 4;
    atomicAdd(ap + 0, hv.x * norm);
    atomicAdd(ap + 1, hv.y * norm);
    atomicAdd(ap + 2, hv.z * norm);
    atomicAdd(ap + 3, hv.w * norm);
}

// ---------------- self-loop + bias + segment-sum pooling ----------------
__global__ void selfloop_bias_pool(const float* __restrict__ Hfeat, const float* __restrict__ agg,
                                   const float* __restrict__ deg, const float* __restrict__ bias,
                                   const int* __restrict__ batch, float* __restrict__ pool,
                                   float* __restrict__ cnt, int addCnt) {
    int idx = blockIdx.x * blockDim.x + threadIdx.x;    // over N*H
    if (idx >= NN * HH) return;
    int node = idx >> 7;
    int f    = idx & 127;
    float v = agg[idx] + Hfeat[idx] / deg[node] + bias[f];  // dinv^2 = 1/deg
    int b = batch[node];
    atomicAdd(&pool[(size_t)b * HH + f], v);
    if (addCnt && f == 0) atomicAdd(&cnt[b], 1.0f);
}

// ---------------- fusion: MHA(S=2,NH=4) + residual + LN + mean + output linear ----------------
// One block per graph, 128 threads (thread = feature channel).
__global__ void fuse_attention(const float* __restrict__ pool_ego, const float* __restrict__ pool_cut,
                               const float* __restrict__ cnt,
                               const float* __restrict__ ipw, const float* __restrict__ ipb,
                               const float* __restrict__ opw, const float* __restrict__ opb,
                               const float* __restrict__ lnw, const float* __restrict__ lnb,
                               const float* __restrict__ ow,  const float* __restrict__ ob,
                               float* __restrict__ out) {
    __shared__ float feat[2][HH];
    __shared__ float q[2][HH], k[2][HH], v[2][HH];
    __shared__ float att[4][2][2];
    __shared__ float o[2][HH];
    __shared__ float y[2][HH];
    __shared__ float red[2][2];     // [s][0]=mean, [s][1]=inv std
    __shared__ float fused[HH];

    const int b = blockIdx.x;
    const int t = threadIdx.x;

    float c = fmaxf(cnt[b], 1.0f);
    feat[0][t] = pool_ego[(size_t)b * HH + t] / c;
    feat[1][t] = pool_cut[(size_t)b * HH + t] / c;
    __syncthreads();

    // qkv projections: in_proj_w is [3H, H]
    for (int s = 0; s < 2; ++s) {
        float aq = ipb[t], ak = ipb[HH + t], av = ipb[2 * HH + t];
        for (int j = 0; j < HH; ++j) {
            float fj = feat[s][j];
            aq += ipw[(size_t)t * HH + j] * fj;
            ak += ipw[(size_t)(HH + t) * HH + j] * fj;
            av += ipw[(size_t)(2 * HH + t) * HH + j] * fj;
        }
        q[s][t] = aq; k[s][t] = ak; v[s][t] = av;
    }
    __syncthreads();

    // scores: 4 heads x (s1,s2), head dim 32; scale 1/sqrt(32)
    if (t < 16) {
        int head = t >> 2, s1 = (t >> 1) & 1, s2 = t & 1;
        float sc = 0.f;
        for (int d = 0; d < 32; ++d) sc += q[s1][head * 32 + d] * k[s2][head * 32 + d];
        att[head][s1][s2] = sc * 0.17677669529663687f;
    }
    __syncthreads();
    // softmax over s2 (2 entries)
    if (t < 8) {
        int head = t >> 1, s1 = t & 1;
        float a0 = att[head][s1][0], a1 = att[head][s1][1];
        float m  = fmaxf(a0, a1);
        float e0 = expf(a0 - m), e1 = expf(a1 - m);
        float inv = 1.f / (e0 + e1);
        att[head][s1][0] = e0 * inv;
        att[head][s1][1] = e1 * inv;
    }
    __syncthreads();

    int head = t >> 5;   // head owning this channel (contiguous 32-chunks)
    for (int s = 0; s < 2; ++s)
        o[s][t] = att[head][s][0] * v[0][t] + att[head][s][1] * v[1][t];
    __syncthreads();

    // out projection + residual
    for (int s = 0; s < 2; ++s) {
        float acc = opb[t];
        for (int j = 0; j < HH; ++j) acc += opw[(size_t)t * HH + j] * o[s][j];
        y[s][t] = acc + feat[s][t];
    }
    __syncthreads();

    // LayerNorm stats per sequence position
    if (t < 2) {
        float m = 0.f;
        for (int j = 0; j < HH; ++j) m += y[t][j];
        m /= (float)HH;
        float var = 0.f;
        for (int j = 0; j < HH; ++j) { float d = y[t][j] - m; var += d * d; }
        var /= (float)HH;
        red[t][0] = m;
        red[t][1] = rsqrtf(var + 1e-5f);
    }
    __syncthreads();

    float l0 = (y[0][t] - red[0][0]) * red[0][1] * lnw[t] + lnb[t];
    float l1 = (y[1][t] - red[1][0]) * red[1][1] * lnw[t] + lnb[t];
    fused[t] = 0.5f * (l0 + l1);
    __syncthreads();

    if (t < OO) {
        float acc = ob[t];
        for (int j = 0; j < HH; ++j) acc += ow[(size_t)t * HH + j] * fused[j];
        out[(size_t)b * OO + t] = acc;
    }
}

// ---------------- launcher ----------------
extern "C" void kernel_launch(void* const* d_in, const int* in_sizes, int n_in,
                              void* d_out, int out_size, void* d_ws, size_t ws_size,
                              hipStream_t stream) {
    const float* x_ego  = (const float*)d_in[0];
    const int*   ei_ego = (const int*)  d_in[1];
    const float* x_cut  = (const float*)d_in[2];
    const int*   ei_cut = (const int*)  d_in[3];
    const int*   batch  = (const int*)  d_in[4];
    const float* W1     = (const float*)d_in[5];
    const float* b1     = (const float*)d_in[6];
    const float* W2     = (const float*)d_in[7];
    const float* b2     = (const float*)d_in[8];
    const float* ipw    = (const float*)d_in[9];
    const float* ipb    = (const float*)d_in[10];
    const float* opw    = (const float*)d_in[11];
    const float* opb    = (const float*)d_in[12];
    const float* lnw    = (const float*)d_in[13];
    const float* lnb    = (const float*)d_in[14];
    const float* ow     = (const float*)d_in[15];
    const float* ob     = (const float*)d_in[16];
    float* out = (float*)d_out;

    // workspace layout (floats)
    float* ws = (float*)d_ws;
    const long NH   = (long)NN * HH;      // 6,400,000
    float* agg_ego  = ws;
    float* agg_cut  = agg_ego + NH;
    float* pool_ego = agg_cut + NH;
    float* pool_cut = pool_ego + (long)BB * HH;
    float* cnt      = pool_cut + (long)BB * HH;
    float* deg_ego  = cnt + 64;
    float* deg_cut  = deg_ego + NN;
    float* h_ego    = deg_cut + NN;
    float* h_cut    = h_ego + NH;

    // 1) init: zero [agg x2, pool x2, cnt]; deg = 1.0 (self loop)
    const long zeroN = 2 * NH + 2L * BB * HH + 64;
    fill_f32<<<(int)((zeroN + 255) / 256), 256, 0, stream>>>(agg_ego, 0.0f, zeroN);
    fill_f32<<<(2 * NN + 255) / 256, 256, 0, stream>>>(deg_ego, 1.0f, 2L * NN);

    // 2) degrees
    degree_kernel<<<(EE + 255) / 256, 256, 0, stream>>>(ei_ego, deg_ego, EE);
    degree_kernel<<<(EE + 255) / 256, 256, 0, stream>>>(ei_cut, deg_cut, EE);

    // 3) GEMMs via WMMA: h = x @ W^T; 3125 row tiles, 8 per block
    const int gblocks = (NN / 16 + 7) / 8;   // 391
    gemm_xwT_wmma<<<gblocks, 256, 0, stream>>>(x_ego, W1, h_ego, NN);
    gemm_xwT_wmma<<<gblocks, 256, 0, stream>>>(x_cut, W2, h_cut, NN);

    // 4) normalized edge scatter (wave per edge)
    const int sblocks = (EE * 32 + 255) / 256;    // 100000
    edge_scatter<<<sblocks, 256, 0, stream>>>(h_ego, ei_ego, deg_ego, agg_ego, EE);
    edge_scatter<<<sblocks, 256, 0, stream>>>(h_cut, ei_cut, deg_cut, agg_cut, EE);

    // 5) self-loop term + bias + pooling sums (+node counts once)
    selfloop_bias_pool<<<(NN * HH) / 256, 256, 0, stream>>>(h_ego, agg_ego, deg_ego, b1,
                                                            batch, pool_ego, cnt, 1);
    selfloop_bias_pool<<<(NN * HH) / 256, 256, 0, stream>>>(h_cut, agg_cut, deg_cut, b2,
                                                            batch, pool_cut, cnt, 0);

    // 6) attention fusion + LN + output
    fuse_attention<<<BB, HH, 0, stream>>>(pool_ego, pool_cut, cnt, ipw, ipb, opw, opb,
                                          lnw, lnb, ow, ob, out);
}
